// EncoderLSTM_88064009437553
// MI455X (gfx1250) — matrix operand order repaired and checked by reading, not compiled, not run on hardware
//
#include <hip/hip_runtime.h>
#include <hip/hip_bf16.h>

typedef __attribute__((ext_vector_type(16))) _Float16 v16h;
typedef __attribute__((ext_vector_type(8)))  float    v8f;

// ---------------------------------------------------------------------------
// Pack OIHW f32 weights into WMMA A-fragment layout, f16, K padded to 32.
// packed[((kb*Mblk + mb)*32 + lane)*16 + j]
//   m = mb*16 + (lane&15), hi = lane>>4
//   k(j) = kb*32 + hi*16 + j      (contiguous k order, matches B LDS read)
// ---------------------------------------------------------------------------
__global__ void pack_weights_kernel(const float* __restrict__ w,
                                    _Float16* __restrict__ packed,
                                    int Cout, int Ktot) {
  const int Mblk = Cout >> 4;
  const int blk  = blockIdx.x;
  const int kb   = blk / Mblk;
  const int mb   = blk - kb * Mblk;
  const int lane = threadIdx.x;
  const int m    = (mb << 4) + (lane & 15);
  const int hi   = lane >> 4;
  v16h frag;
#pragma unroll
  for (int j = 0; j < 16; ++j) {
    int k = (kb << 5) + (hi << 4) + j;
    float v = (k < Ktot) ? w[(size_t)m * Ktot + k] : 0.0f;
    frag[j] = (_Float16)v;
  }
  *(v16h*)(packed + ((size_t)blk * 32 + lane) * 16) = frag;
}

// ---------------------------------------------------------------------------
// Implicit-GEMM convolution via v_wmma_f32_16x16x32_f16.
// Block = 128 threads (4 waves). Block tile = 64 couts x 64 output pixels.
// Geometry is compile-time -> all im2col decode divisions are
// constant-divisor (mul_hi+shift), no runtime rcp chains.
// LDS B tile is double-buffered: gather of K-block kb+1 overlaps the 4 WMMAs
// of K-block kb; one barrier per K-block.
// ---------------------------------------------------------------------------
template <typename TIN, bool RELU, bool OUT16,
          int H, int W, int OH, int OW, int KH, int KW, int STRIDE, int PAD>
__global__ __launch_bounds__(128)
void conv_wmma_kernel(const TIN* __restrict__ inA, int Ca, long long sA,
                      const _Float16* __restrict__ inB, long long sB,
                      const _Float16* __restrict__ packedW,
                      const float* __restrict__ bias,
                      float* __restrict__ out32, _Float16* __restrict__ out16,
                      int NB, int Cout, int Ktot, int nKblk) {
  constexpr int OHOW = OH * OW;
  constexpr int KHKW = KH * KW;
  const int tid     = threadIdx.x;
  const int lane    = tid & 31;
  const int wave    = tid >> 5;
  const int mb      = blockIdx.y * 4 + wave;     // 16-row cout block
  const int MblkTot = Cout >> 4;
  const int n0      = blockIdx.x << 6;           // 64 output pixels
  const int N       = NB * OHOW;

  __shared__ _Float16 Bt[2][64 * 32] __attribute__((aligned(64)));  // [n][k]

  // Gather assignment: thread covers k = tid&31 (invariant) and
  // n = (tid>>5) + 4*i, i = 0..15. Pre-decode the 16 pixel coords once.
  const int k     = tid & 31;
  const int nbase = tid >> 5;
  int nb[16], ihb[16], iwb[16];
#pragma unroll
  for (int i = 0; i < 16; ++i) {
    const int gn = n0 + nbase + (i << 2);
    if (gn < N) {
      const int b  = gn / OHOW;            // constant divisor
      const int sp = gn - b * OHOW;
      const int oh = sp / OW;              // constant divisor
      const int ow = sp - oh * OW;
      nb[i]  = b;
      ihb[i] = oh * STRIDE - PAD;
      iwb[i] = ow * STRIDE - PAD;
    } else {
      nb[i] = -1; ihb[i] = 0; iwb[i] = 0;
    }
  }

  // Cooperative im2col gather of one 32K x 64N tile (16 elems/thread).
  auto gather = [&](int kb, _Float16* __restrict__ dst) {
    const int gk = (kb << 5) + k;
    const bool kvalid = gk < Ktot;
    int cin = 0, kh = 0, kw = 0;
    if (kvalid) {
      cin = gk / KHKW;                     // constant divisor
      const int r = gk - cin * KHKW;
      kh = r / KW;                         // constant divisor
      kw = r - kh * KW;
    }
#pragma unroll
    for (int i = 0; i < 16; ++i) {
      _Float16 v = (_Float16)0.0f;
      const int ih = ihb[i] + kh;
      const int iw = iwb[i] + kw;
      if (kvalid && nb[i] >= 0 &&
          (unsigned)ih < (unsigned)H && (unsigned)iw < (unsigned)W) {
        if (cin < Ca)
          v = (_Float16)inA[(size_t)nb[i] * sA + ((size_t)cin * H + ih) * W + iw];
        else
          v = inB[(size_t)nb[i] * sB + ((size_t)(cin - Ca) * H + ih) * W + iw];
      }
      dst[((nbase + (i << 2)) << 5) + k] = v;
    }
  };

  v8f acc[4] = {};

  gather(0, Bt[0]);
  __syncthreads();

  for (int kb = 0; kb < nKblk; ++kb) {
    const int cur = kb & 1;
    // prefetch-gather next K-block into the other buffer (overlaps WMMAs)
    if (kb + 1 < nKblk) gather(kb + 1, Bt[cur ^ 1]);

    // A fragment: one contiguous 32B load from the packed-weight panel.
    const v16h afrag =
        *(const v16h*)(packedW + (((size_t)kb * MblkTot + mb) * 32 + lane) * 16);
    if (kb + 1 < nKblk)
      __builtin_prefetch(
          packedW + (((size_t)(kb + 1) * MblkTot + mb) * 32 + lane) * 16, 0, 1);

    // 4 N-subtiles against one A fragment.
#pragma unroll
    for (int s = 0; s < 4; ++s) {
      const v16h bfrag =
          *(const v16h*)(Bt[cur] + ((((s << 4) + (lane & 15))) << 5) +
                         ((lane >> 4) << 4));
      acc[s] = __builtin_amdgcn_wmma_f32_16x16x32_f16(false, afrag, false, bfrag,
                                                      (short)0, acc[s], false,
                                                      false);
    }
    __syncthreads();   // readers of Bt[cur] done; writers of Bt[cur^1] done
  }

  // Epilogue. C layout: VGPR r holds (M = 8*hi + r, N = lane&15).
  const int m0 = (mb << 4) + ((lane >> 4) << 3);
#pragma unroll
  for (int s = 0; s < 4; ++s) {
    const int gn = n0 + (s << 4) + (lane & 15);
    if (gn < N) {
      const int b  = gn / OHOW;
      const int sp = gn - b * OHOW;
#pragma unroll
      for (int r2 = 0; r2 < 8; ++r2) {
        const int m = m0 + r2;
        float v = acc[s][r2] + bias[m];
        if (RELU) v = fmaxf(v, 0.0f);
        const size_t oidx = ((size_t)b * Cout + m) * OHOW + sp;
        if (OUT16) out16[oidx] = (_Float16)v;
        else       out32[oidx] = v;
      }
    }
  }
}

// ---------------------------------------------------------------------------
// ConvLSTM pointwise: gates (i,f,o,g) -> c,h update. h stored f16 for the
// next step's WMMA conv and into the (B,T,hid,HW) sequence buffer.
// ---------------------------------------------------------------------------
__global__ void lstm_point_kernel(const float* __restrict__ gates,
                                  float* __restrict__ c,
                                  _Float16* __restrict__ h16,
                                  _Float16* __restrict__ hseq,
                                  int B_, int hid, int HW, int t, int T_,
                                  int first) {
  const int idx   = blockIdx.x * blockDim.x + threadIdx.x;
  const int total = B_ * hid * HW;
  if (idx >= total) return;
  const int b   = idx / (hid * HW);
  const int rem = idx - b * hid * HW;
  const int ch  = rem / HW;
  const int sp  = rem - ch * HW;
  const size_t gbase = (size_t)b * 4 * hid * HW;
  float iv = gates[gbase + (size_t)ch * HW + sp];
  float fv = gates[gbase + (size_t)(hid + ch) * HW + sp];
  float ov = gates[gbase + (size_t)(2 * hid + ch) * HW + sp];
  float gv = gates[gbase + (size_t)(3 * hid + ch) * HW + sp];
  iv = 1.0f / (1.0f + __expf(-iv));
  fv = 1.0f / (1.0f + __expf(-fv));
  ov = 1.0f / (1.0f + __expf(-ov));
  gv = tanhf(gv);
  const float cp = first ? 0.0f : c[idx];
  const float cn = fv * cp + iv * gv;
  const float hn = ov * tanhf(cn);
  c[idx]   = cn;
  h16[idx] = (_Float16)hn;
  hseq[(((size_t)b * T_ + t) * hid + ch) * HW + sp] = (_Float16)hn;
}

// ---------------------------------------------------------------------------
// BatchNorm (training mode, batch statistics): one block per channel.
// ---------------------------------------------------------------------------
template <typename T>
__global__ void bn_stats_kernel(const T* __restrict__ x, float* __restrict__ mean,
                                float* __restrict__ rstd, int BT, int C, int HW) {
  const int c   = blockIdx.x;
  const int tid = threadIdx.x;
  float s = 0.0f, s2 = 0.0f;
  const int total = BT * HW;
  for (int i = tid; i < total; i += blockDim.x) {
    const int bt = i / HW;
    const int sp = i - bt * HW;
    const float v = (float)x[((size_t)bt * C + c) * HW + sp];
    s += v;
    s2 += v * v;
  }
  __shared__ float sh[256], sh2[256];
  sh[tid] = s;
  sh2[tid] = s2;
  __syncthreads();
  for (int off = 128; off > 0; off >>= 1) {
    if (tid < off) {
      sh[tid] += sh[tid + off];
      sh2[tid] += sh2[tid + off];
    }
    __syncthreads();
  }
  if (tid == 0) {
    const float inv = 1.0f / (float)total;
    const float m   = sh[0] * inv;
    const float var = sh2[0] * inv - m * m;
    mean[c] = m;
    rstd[c] = rsqrtf(var + 1e-5f);
  }
}

template <typename TIN, typename TOUT, bool RELU>
__global__ void bn_apply_kernel(const TIN* __restrict__ x, TOUT* __restrict__ y,
                                const float* __restrict__ mean,
                                const float* __restrict__ rstd,
                                const float* __restrict__ g,
                                const float* __restrict__ bb,
                                int C, int HW, long long total) {
  const long long idx = (long long)blockIdx.x * blockDim.x + threadIdx.x;
  if (idx >= total) return;
  const int c = (int)((idx / HW) % C);
  float v = ((float)x[idx] - mean[c]) * rstd[c] * g[c] + bb[c];
  if (RELU) v = fmaxf(v, 0.0f);
  y[idx] = (TOUT)v;
}

__global__ void zero_f16_kernel(_Float16* __restrict__ p, int n) {
  const int i = blockIdx.x * blockDim.x + threadIdx.x;
  if (i < n) p[i] = (_Float16)0.0f;
}

// ---------------------------------------------------------------------------
extern "C" void kernel_launch(void* const* d_in, const int* in_sizes, int n_in,
                              void* d_out, int out_size, void* d_ws, size_t ws_size,
                              hipStream_t stream) {
  (void)in_sizes; (void)n_in; (void)out_size; (void)ws_size;
  const float* x       = (const float*)d_in[0];
  const float* conv1_w = (const float*)d_in[1];
  const float* conv1_b = (const float*)d_in[2];
  const float* conv2_w = (const float*)d_in[3];
  const float* conv2_b = (const float*)d_in[4];
  const float* bn0_g   = (const float*)d_in[5];
  const float* bn0_b   = (const float*)d_in[6];
  const float* lstm1_w = (const float*)d_in[7];
  const float* lstm1_b = (const float*)d_in[8];
  const float* bn1_g   = (const float*)d_in[9];
  const float* bn1_b   = (const float*)d_in[10];
  const float* lstm2_w = (const float*)d_in[11];
  const float* lstm2_b = (const float*)d_in[12];
  const float* bn2_g   = (const float*)d_in[13];
  const float* bn2_b   = (const float*)d_in[14];

  constexpr int B = 8, T = 16, BT = 128;

  char* wsb = (char*)d_ws;
  size_t off = 0;
  auto alloc = [&](size_t bytes) -> void* {
    void* p = wsb + off;
    off += (bytes + 255) & ~(size_t)255;
    return p;
  };

  _Float16* pw1    = (_Float16*)alloc((size_t)12 * 8 * 512 * 2);   // conv1 K=363->384
  _Float16* pw2    = (_Float16*)alloc((size_t)100 * 4 * 512 * 2);  // conv2 K=3200
  _Float16* pw3    = (_Float16*)alloc((size_t)27 * 8 * 512 * 2);   // lstm1 K=864
  _Float16* pw4    = (_Float16*)alloc((size_t)14 * 4 * 512 * 2);   // lstm2 K=432
  _Float16* act1   = (_Float16*)alloc((size_t)BT * 128 * 3136 * 2);
  float*    act2   = (float*)   alloc((size_t)BT * 64 * 784 * 4);
  _Float16* bn0out = (_Float16*)alloc((size_t)BT * 64 * 784 * 2);
  float*    gates  = (float*)   alloc((size_t)B * 128 * 784 * 4);
  float*    c1     = (float*)   alloc((size_t)B * 32 * 784 * 4);
  _Float16* h1     = (_Float16*)alloc((size_t)B * 32 * 784 * 2);
  _Float16* hseq1  = (_Float16*)alloc((size_t)B * T * 32 * 784 * 2);
  _Float16* bn1out = (_Float16*)alloc((size_t)B * T * 32 * 784 * 2);
  float*    c2     = (float*)   alloc((size_t)B * 16 * 784 * 4);
  _Float16* h2     = (_Float16*)alloc((size_t)B * 16 * 784 * 2);
  _Float16* hseq2  = (_Float16*)alloc((size_t)B * T * 16 * 784 * 2);
  float*    bmean  = (float*)alloc(128 * 4);
  float*    brstd  = (float*)alloc(128 * 4);

  // ---- pack weights to WMMA fragment layout (f16) ----
  pack_weights_kernel<<<12 * 8, 32, 0, stream>>>(conv1_w, pw1, 128, 363);
  pack_weights_kernel<<<100 * 4, 32, 0, stream>>>(conv2_w, pw2, 64, 3200);
  pack_weights_kernel<<<27 * 8, 32, 0, stream>>>(lstm1_w, pw3, 128, 864);
  pack_weights_kernel<<<14 * 4, 32, 0, stream>>>(lstm2_w, pw4, 64, 432);

  // ---- conv1 (11x11 s4 p4) + bias + ReLU -> f16 ----
  conv_wmma_kernel<float, true, true, 224, 224, 56, 56, 11, 11, 4, 4>
      <<<dim3(6272, 2), 128, 0, stream>>>(
          x, 3, (long long)3 * 224 * 224, nullptr, 0,
          pw1, conv1_b, nullptr, act1, BT, 128, 363, 12);

  // ---- conv2 (5x5 s2 p2) + bias -> f32 (for BN stats) ----
  conv_wmma_kernel<_Float16, false, false, 56, 56, 28, 28, 5, 5, 2, 2>
      <<<dim3(1568, 1), 128, 0, stream>>>(
          act1, 128, (long long)128 * 3136, nullptr, 0,
          pw2, conv2_b, act2, nullptr, BT, 64, 3200, 100);

  // ---- BN0 + ReLU -> f16 ----
  bn_stats_kernel<float><<<64, 256, 0, stream>>>(act2, bmean, brstd, BT, 64, 784);
  {
    const long long total = (long long)BT * 64 * 784;
    bn_apply_kernel<float, _Float16, true>
        <<<(int)((total + 255) / 256), 256, 0, stream>>>(
            act2, bn0out, bmean, brstd, bn0_g, bn0_b, 64, 784, total);
  }

  // ---- ConvLSTM 1 (hidden 32) ----
  zero_f16_kernel<<<(B * 32 * 784 + 255) / 256, 256, 0, stream>>>(h1, B * 32 * 784);
  for (int t = 0; t < T; ++t) {
    conv_wmma_kernel<_Float16, false, false, 28, 28, 28, 28, 3, 3, 1, 1>
        <<<dim3(98, 2), 128, 0, stream>>>(
            bn0out + (size_t)t * 64 * 784, 64, (long long)T * 64 * 784,
            h1, (long long)32 * 784,
            pw3, lstm1_b, gates, nullptr, B, 128, 864, 27);
    lstm_point_kernel<<<(B * 32 * 784 + 255) / 256, 256, 0, stream>>>(
        gates, c1, h1, hseq1, B, 32, 784, t, T, t == 0 ? 1 : 0);
  }
  bn_stats_kernel<_Float16><<<32, 256, 0, stream>>>(hseq1, bmean, brstd, BT, 32, 784);
  {
    const long long total = (long long)BT * 32 * 784;
    bn_apply_kernel<_Float16, _Float16, false>
        <<<(int)((total + 255) / 256), 256, 0, stream>>>(
            hseq1, bn1out, bmean, brstd, bn1_g, bn1_b, 32, 784, total);
  }

  // ---- ConvLSTM 2 (hidden 16) ----
  zero_f16_kernel<<<(B * 16 * 784 + 255) / 256, 256, 0, stream>>>(h2, B * 16 * 784);
  for (int t = 0; t < T; ++t) {
    conv_wmma_kernel<_Float16, false, false, 28, 28, 28, 28, 3, 3, 1, 1>
        <<<dim3(98, 1), 128, 0, stream>>>(
            bn1out + (size_t)t * 32 * 784, 32, (long long)T * 32 * 784,
            h2, (long long)16 * 784,
            pw4, lstm2_b, gates, nullptr, B, 64, 432, 14);
    lstm_point_kernel<<<(B * 16 * 784 + 255) / 256, 256, 0, stream>>>(
        gates, c2, h2, hseq2, B, 16, 784, t, T, t == 0 ? 1 : 0);
  }

  // ---- BN2 -> f32 straight into d_out (B, T*16*28*28 contiguous) ----
  bn_stats_kernel<_Float16><<<16, 256, 0, stream>>>(hseq2, bmean, brstd, BT, 16, 784);
  {
    const long long total = (long long)BT * 16 * 784;
    bn_apply_kernel<_Float16, float, false>
        <<<(int)((total + 255) / 256), 256, 0, stream>>>(
            hseq2, (float*)d_out, bmean, brstd, bn2_g, bn2_b, 16, 784, total);
  }
}
